// CustomSelfAttention_22479858827522
// MI455X (gfx1250) — matrix operand order repaired and checked by reading, not compiled, not run on hardware
//
#include <hip/hip_runtime.h>

typedef __bf16 bf16;
typedef __attribute__((ext_vector_type(16))) __bf16 v16bf;
typedef __attribute__((ext_vector_type(8)))  __bf16 v8bf;
typedef __attribute__((ext_vector_type(4)))  __bf16 v4bf;
typedef __attribute__((ext_vector_type(8)))  float  v8f;
typedef __attribute__((ext_vector_type(4)))  float  v4f;
typedef __attribute__((ext_vector_type(4)))  unsigned int u32x4;
typedef __attribute__((ext_vector_type(4)))  int  i32x4;
typedef __attribute__((ext_vector_type(8)))  int  i32x8;

#define S_LEN 1024
#define HEADS 16
#define HDIM  64
#define DIST_ROWS 2047   // 2*MAX_POS-1
#define CB_LD 192

// attention kernel dynamic-LDS layout (byte offsets from smem[0])
#define PLDS_OFF  (CB_LD*CB_LD*2)          // cband bf16 [192][192]   73728
#define KLDS_OFF  (PLDS_OFF + 8*16*64*2)   // P staging  [8][16][64]  90112
#define VLDS_OFF  (KLDS_OFF + 64*64*2)     // K tile     [64][64]     98304
#define ATTN_SMEM (VLDS_OFF + 64*64*2)     // V^T tile   [64][64]    106496

#if defined(__has_builtin)
# if __has_builtin(__builtin_amdgcn_tensor_load_to_lds)
#  define HAVE_TDM 1
# endif
#endif

// ---------------- fragment helpers (WMMA 16x16x32 bf16, k order used by this code) --------
// A (16x32, MxK): lane m=L&15 holds row m; 16 contiguous k at k = 16*hi
// (hi = L>>4), loaded as two 8-elt chunks.   (A and B share it; not the ISA table order)
// B (32x16, KxN): lane n=L&15 holds column n; 16 contiguous k at k = 16*hi.
// So with B^T stored row-major, a B fragment is one 32B contiguous load/lane.
// C/D (16x16 f32): lane L element i -> (row i+8*hi, col L&15).

__device__ __forceinline__ v8bf ld8bf(const bf16* p){ return *(const v8bf*)p; }

__device__ __forceinline__ v16bf cat16(v8bf a, v8bf b){
  union { v16bf v; v8bf h[2]; } u; u.h[0] = a; u.h[1] = b; return u.v;
}
__device__ __forceinline__ v16bf a_frag(const bf16* row_ptr, int hi){
  return cat16(ld8bf(row_ptr + 16*hi), ld8bf(row_ptr + 16*hi + 8));
}
__device__ __forceinline__ v8f wmma_bf16(v16bf a, v16bf b, v8f c){
  return __builtin_amdgcn_wmma_f32_16x16x32_bf16(false, a, false, b,
                                                 (short)0, c, false, false);
}

#ifdef HAVE_TDM
// ---- Tensor Data Mover: 2-D tile (rows x cols of 2-byte elems) -> LDS ------
// D# per CDNA5 ISA ch.8: group0 = {count/flags, lds_addr, global_addr, type=2},
// group1 = {mask/data_size, tensor_dim0/1, tile_dim0/1/2, dim0_stride, ...}.
__device__ __forceinline__ void tdm_load_2d(unsigned lds_off, const void* gaddr,
    unsigned dim0, unsigned dim1, unsigned tile0, unsigned tile1,
    unsigned stride0)
{
  unsigned long long ga = (unsigned long long)(__SIZE_TYPE__)gaddr;
  u32x4 g0;
  g0[0] = 1u;                                    // count=1, user descriptor
  g0[1] = lds_off;                               // lds_addr (bytes)
  g0[2] = (unsigned)(ga & 0xffffffffu);          // global_addr[31:0]
  g0[3] = (unsigned)((ga >> 32) & 0x01ffffffu)   // global_addr[56:32]
        | (2u << 30);                            // type = 2 ("image")
  i32x8 g1;
  g1[0] = (int)(1u << 16);                       // data_size=1 (2B), mask=0
  g1[1] = (int)((dim0 & 0xffffu) << 16);         // tensor_dim0[15:0]
  g1[2] = (int)(((dim0 >> 16) & 0xffffu) | ((dim1 & 0xffffu) << 16));
  g1[3] = (int)(((dim1 >> 16) & 0xffffu) | ((tile0 & 0xffffu) << 16));
  g1[4] = (int)(tile1 & 0xffffu);                // tile_dim1 (tile_dim2 = 0)
  g1[5] = (int)stride0;                          // tensor_dim0_stride[31:0]
  g1[6] = 0;
  g1[7] = 0;
  i32x4 z4 = {0, 0, 0, 0};
#if defined(__clang_major__) && (__clang_major__ >= 23)
  i32x8 z8 = {0, 0, 0, 0, 0, 0, 0, 0};
  __builtin_amdgcn_tensor_load_to_lds(g0, g1, z4, z4, z8, 0);
#else
  __builtin_amdgcn_tensor_load_to_lds(g0, g1, z4, z4, 0);
#endif
}
#endif

// ---------------- kernel 0: fp32 -> bf16 bulk convert (4 elems / thread) ----
__global__ void cvt_bf16_kernel(const float* __restrict__ src,
                                bf16* __restrict__ dst, int n4){
  int i = blockIdx.x * 256 + threadIdx.x;
  if (i < n4){
    v4f f = ((const v4f*)src)[i];
    v4bf o;
    #pragma unroll
    for (int j = 0; j < 4; ++j) o[j] = (bf16)f[j];
    ((v4bf*)dst)[i] = o;
  }
}

// ---------------- kernel 1: fused QKV projection (Y = X*W^T + b) ------------
// grid (32 Mtiles, 16 Ntiles, 3 proj), block 256 = 8 waves; wave -> 16x64 tile.
__global__ void __launch_bounds__(256) qkv_kernel(
    const bf16* __restrict__ xb,   // [4096,1024] bf16 hidden
    const bf16* __restrict__ wb,   // [3,1024,1024] bf16 Wq|Wk|Wv
    const float* __restrict__ bq, const float* __restrict__ bk,
    const float* __restrict__ bv,
    bf16* __restrict__ q, bf16* __restrict__ k, bf16* __restrict__ vt)
{
  const bf16* W = wb + (size_t)blockIdx.z * (1024*1024);
  const float* bias; bf16* out; bool tr;
  if      (blockIdx.z == 0){ bias = bq; out = q;  tr = false; }
  else if (blockIdx.z == 1){ bias = bk; out = k;  tr = false; }
  else                     { bias = bv; out = vt; tr = true;  }

  const int lane = threadIdx.x & 31;
  const int wave = threadIdx.x >> 5;
  const int hi = lane >> 4;
  const int lm = lane & 15;

  const int m0 = blockIdx.x * 128 + wave * 16;  // output row base (B*S flat)
  const int n0 = blockIdx.y * 64;               // output col base (D)

  v8f acc[4];
  #pragma unroll
  for (int nf = 0; nf < 4; ++nf)
    #pragma unroll
    for (int i = 0; i < 8; ++i) acc[nf][i] = 0.f;

  const bf16* xrow = xb + (m0 + lm) * 1024;         // this lane's A row
  const bf16* wrow0 = W + (n0 + lm) * 1024 + 16*hi; // this lane's B^T row base
  for (int kk = 0; kk < 1024; kk += 32){
    v16bf a = a_frag(xrow + kk, hi);
    #pragma unroll
    for (int nf = 0; nf < 4; ++nf){
      v16bf bfr = *(const v16bf*)(wrow0 + nf*(16*1024) + kk);
      acc[nf] = wmma_bf16(a, bfr, acc[nf]);
    }
  }

  #pragma unroll
  for (int nf = 0; nf < 4; ++nf){
    int n  = n0 + nf*16 + lm;
    float bb = bias[n];
    int h = n >> 6, hd = n & 63;
    #pragma unroll
    for (int i = 0; i < 8; ++i){
      int m = m0 + i + 8*hi;
      int b = m >> 10, s = m & 1023;
      float y = acc[nf][i] + bb;
      if (!tr) out[(((b*HEADS + h)*S_LEN) + s)*HDIM + hd] = (bf16)y;  // [BH,S,HD]
      else     out[(((b*HEADS + h)*HDIM) + hd)*S_LEN + s] = (bf16)y;  // [BH,HD,S]
    }
  }
}

// ---------------- kernel 2: fused flash attention + relative position bias --
// grid (8 query tiles, 64 b*h), block 256 = 8 waves, wave owns 16 query rows.
__global__ void __launch_bounds__(256) attn_kernel(
    const bf16* __restrict__ q, const bf16* __restrict__ k,
    const bf16* __restrict__ vt, const bf16* __restrict__ dist,
    const float* __restrict__ amask, float* __restrict__ out)
{
  extern __shared__ char smem[];
  bf16* cband = (bf16*)smem;                 // [192][192] band GEMM out (bf16)
  bf16* plds  = (bf16*)(smem + PLDS_OFF);    // [8][16][64] P staging
  bf16* klds  = (bf16*)(smem + KLDS_OFF);    // [64][64] K tile (rows k0..k0+63)
  bf16* vlds  = (bf16*)(smem + VLDS_OFF);    // [64][64] V^T tile (hd x key)

  const int lane = threadIdx.x & 31;
  const int wave = threadIdx.x >> 5;
  const int hi = lane >> 4;
  const int lm = lane & 15;

  const int bh = blockIdx.y;            // b*16 + h
  const int b  = bh >> 4;
  const int h  = bh & 15;
  const int l0 = blockIdx.x * 128;

  const bf16* qh = q  + (size_t)bh * (S_LEN*HDIM);
  const bf16* kh = k  + (size_t)bh * (S_LEN*HDIM);
  const bf16* vh = vt + (size_t)bh * (HDIM*S_LEN);

  // Q A-fragments for this wave's 16 rows, hd 0..31 / 32..63 (held all kernel)
  const bf16* qrow = qh + (l0 + wave*16 + lm) * HDIM;
  v16bf qa0 = a_frag(qrow, hi);
  v16bf qa1 = a_frag(qrow + 32, hi);

  float mstate[8], lstate[8];
  v8f acc[4];
  #pragma unroll
  for (int i = 0; i < 8; ++i){ mstate[i] = -1e30f; lstate[i] = 0.f; }
  #pragma unroll
  for (int hf = 0; hf < 4; ++hf)
    #pragma unroll
    for (int i = 0; i < 8; ++i) acc[hf][i] = 0.f;

  bf16* pwave = plds + wave * (16*64);

  #pragma unroll 1
  for (int k0 = 0; k0 < S_LEN; k0 += 64){
    // Prefetch the next key block into GL2 (gfx1250 global_prefetch_b8).
    {
      int k0n = (k0 + 64 < S_LEN) ? (k0 + 64) : k0;
      __builtin_prefetch(kh + (k0n + lane) * HDIM, 0, 3);
      __builtin_prefetch(kh + (k0n + 32 + lane) * HDIM, 0, 3);
      __builtin_prefetch(vh + lane * S_LEN + k0n, 0, 3);
      __builtin_prefetch(vh + (32 + lane) * S_LEN + k0n, 0, 3);
    }

    // ---- stage K tile [64 keys x 64 hd] and V^T tile [64 hd x 64 keys] in LDS
#ifdef HAVE_TDM
    if (wave == 0){
      tdm_load_2d(KLDS_OFF, kh + (size_t)k0 * HDIM, 64, S_LEN, 64, 64, 64);
      tdm_load_2d(VLDS_OFF, vh + k0, S_LEN, HDIM, 64, 64, S_LEN);
      __builtin_amdgcn_s_wait_tensorcnt(0);
    }
#else
    {
      const bf16* ksrc = kh + (size_t)k0 * HDIM;   // contiguous 8KB
      #pragma unroll
      for (int t2 = (int)threadIdx.x; t2 < 512; t2 += 256)
        *(v8bf*)(klds + t2*8) = *(const v8bf*)(ksrc + t2*8);
      #pragma unroll
      for (int t2 = (int)threadIdx.x; t2 < 512; t2 += 256){
        int row = t2 >> 3, c8 = t2 & 7;
        *(v8bf*)(vlds + row*64 + c8*8) =
            *(const v8bf*)(vh + row*S_LEN + k0 + c8*8);
      }
    }
#endif
    __syncthreads();   // K/V tiles visible to all waves

    // ---- band GEMM: cband[row][t] = A_stacked[row] . dist[tbase+t]
    // rows 0..127 = Q rows l0..l0+127 ; rows 128..191 = K rows k0..k0+63
    const int tbase = l0 - k0 + 960;   // (l0 - (k0+63)) + (MAX_POS-1)
    #pragma unroll 1
    for (int f = wave; f < 144; f += 8){
      int mf = f / 12, tf = f % 12;
      int m = mf*16 + lm;
      int tg = tbase + tf*16 + lm; if (tg > DIST_ROWS - 1) tg = DIST_ROWS - 1;
      const bf16* brow = dist + tg * HDIM;   // B^T row: contiguous d
      v16bf b0 = *(const v16bf*)(brow + 16*hi);
      v16bf b1 = *(const v16bf*)(brow + 32 + 16*hi);
      v16bf a0, a1;
      if (mf < 8){ const bf16* p = qh + (l0 + m) * HDIM;   // Q rows: global
                   a0 = a_frag(p, hi); a1 = a_frag(p + 32, hi); }
      else       { const bf16* p = klds + (m - 128) * HDIM; // K rows: LDS tile
                   a0 = a_frag(p, hi); a1 = a_frag(p + 32, hi); }
      v8f c;
      #pragma unroll
      for (int i = 0; i < 8; ++i) c[i] = 0.f;
      c = wmma_bf16(a0, b0, c);
      c = wmma_bf16(a1, b1, c);
      #pragma unroll
      for (int i = 0; i < 8; ++i)
        cband[(mf*16 + i + 8*hi)*CB_LD + tf*16 + lm] = (bf16)c[i];
    }
    __syncthreads();

    // ---- scores: S = Q K^T  (+ gathered position bias, scale, mask)
    v8f sfr[4];
    #pragma unroll
    for (int nf = 0; nf < 4; ++nf){
      const bf16* krow = klds + (nf*16 + lm) * HDIM;  // B^T row = K row (LDS)
      v8f s;
      #pragma unroll
      for (int i = 0; i < 8; ++i) s[i] = 0.f;
      s = wmma_bf16(qa0, *(const v16bf*)(krow + 16*hi),      s);
      s = wmma_bf16(qa1, *(const v16bf*)(krow + 32 + 16*hi), s);
      #pragma unroll
      for (int i = 0; i < 8; ++i){
        int mb = wave*16 + i + 8*hi;       // row within 128-query tile
        int nb = nf*16 + lm;               // key within 64-key block
        int t  = mb - nb + 63;             // 0..190
        float cb = (float)cband[mb*CB_LD + t]
                 + (float)cband[(128 + nb)*CB_LD + t];
        s[i] = (s[i] + cb) * 0.125f + amask[b*S_LEN + k0 + nb];
      }
      sfr[nf] = s;
    }

    // ---- online softmax (row stats across the 16 lanes of each half-wave)
    float alpha[8];
    #pragma unroll
    for (int i = 0; i < 8; ++i){
      float v = fmaxf(fmaxf(sfr[0][i], sfr[1][i]), fmaxf(sfr[2][i], sfr[3][i]));
      v = fmaxf(v, __shfl_xor(v, 1));
      v = fmaxf(v, __shfl_xor(v, 2));
      v = fmaxf(v, __shfl_xor(v, 4));
      v = fmaxf(v, __shfl_xor(v, 8));
      float mn = fmaxf(mstate[i], v);
      alpha[i] = __expf(mstate[i] - mn);
      mstate[i] = mn;
    }
    float rsum[8];
    #pragma unroll
    for (int i = 0; i < 8; ++i) rsum[i] = 0.f;
    #pragma unroll
    for (int nf = 0; nf < 4; ++nf)
      #pragma unroll
      for (int i = 0; i < 8; ++i){
        float p = __expf(sfr[nf][i] - mstate[i]);
        sfr[nf][i] = p;
        rsum[i] += p;
      }
    #pragma unroll
    for (int i = 0; i < 8; ++i){
      float r = rsum[i];
      r += __shfl_xor(r, 1); r += __shfl_xor(r, 2);
      r += __shfl_xor(r, 4); r += __shfl_xor(r, 8);
      lstate[i] = lstate[i]*alpha[i] + r;
    }

    // ---- stage P (C-layout) -> LDS -> A-layout; DS ops are in-order per wave
    #pragma unroll
    for (int nf = 0; nf < 4; ++nf)
      #pragma unroll
      for (int i = 0; i < 8; ++i)
        pwave[(i + 8*hi)*64 + nf*16 + lm] = (bf16)sfr[nf][i];

    const bf16* prow = pwave + lm*64;
    v16bf pa0 = a_frag(prow, hi);        // keys 0..31 of this block
    v16bf pa1 = a_frag(prow + 32, hi);   // keys 32..63

    // ---- acc = acc*alpha + P V
    #pragma unroll
    for (int hf = 0; hf < 4; ++hf){
      #pragma unroll
      for (int i = 0; i < 8; ++i) acc[hf][i] *= alpha[i];
      const bf16* vrow = vlds + (hf*16 + lm) * 64;  // B^T row = V^T row (LDS)
      acc[hf] = wmma_bf16(pa0, *(const v16bf*)(vrow + 16*hi),      acc[hf]);
      acc[hf] = wmma_bf16(pa1, *(const v16bf*)(vrow + 32 + 16*hi), acc[hf]);
    }
    __syncthreads();   // all reads done before next block's staging/band write
  }

  // ---- epilogue: out[b, l, h*64+hd] = acc / l
  #pragma unroll
  for (int hf = 0; hf < 4; ++hf)
    #pragma unroll
    for (int i = 0; i < 8; ++i){
      int l  = l0 + wave*16 + i + 8*hi;
      int hd = hf*16 + lm;
      out[((size_t)b*S_LEN + l)*1024 + h*HDIM + hd] = acc[hf][i] / lstate[i];
    }
}

// ---------------- host launch ----------------------------------------------
extern "C" void kernel_launch(void* const* d_in, const int* in_sizes, int n_in,
                              void* d_out, int out_size, void* d_ws, size_t ws_size,
                              hipStream_t stream) {
  (void)in_sizes; (void)n_in; (void)out_size; (void)ws_size;
  const float* hidden = (const float*)d_in[0];
  const float* amask  = (const float*)d_in[1];
  // d_in[2] gabriel_mask: dead in the reference (never affects ctx)
  const float* Wq = (const float*)d_in[3];
  const float* bq = (const float*)d_in[4];
  const float* Wk = (const float*)d_in[5];
  const float* bk = (const float*)d_in[6];
  const float* Wv = (const float*)d_in[7];
  const float* bv = (const float*)d_in[8];
  const float* de = (const float*)d_in[9];

  char* ws = (char*)d_ws;
  bf16* xb  = (bf16*)(ws + (size_t) 0u);          // [4096,1024] bf16    8 MB
  bf16* wbb = (bf16*)(ws + (size_t)( 8u << 20));  // [3,1024,1024] bf16  6 MB
  bf16* qb  = (bf16*)(ws + (size_t)(16u << 20));  // [B*H, S, HD]        8 MB
  bf16* kb  = (bf16*)(ws + (size_t)(24u << 20));  // [B*H, S, HD]        8 MB
  bf16* vtb = (bf16*)(ws + (size_t)(32u << 20));  // [B*H, HD, S]        8 MB
  bf16* db  = (bf16*)(ws + (size_t)(40u << 20));  // [2047, 64]        256 KB

  auto cvt = [&](const float* s, bf16* d, int n){
    cvt_bf16_kernel<<<(n/4 + 255)/256, 256, 0, stream>>>(s, d, n/4);
  };
  cvt(hidden, xb, 4096*1024);
  cvt(Wq, wbb + 0*(1024*1024), 1024*1024);
  cvt(Wk, wbb + 1*(1024*1024), 1024*1024);
  cvt(Wv, wbb + 2*(1024*1024), 1024*1024);
  cvt(de, db, DIST_ROWS*HDIM);

  qkv_kernel<<<dim3(32, 16, 3), 256, 0, stream>>>(xb, wbb, bq, bk, bv,
                                                  qb, kb, vtb);
  attn_kernel<<<dim3(8, 64), 256, (size_t)ATTN_SMEM, stream>>>(
      qb, kb, vtb, db, amask, (float*)d_out);
}
